// GAT_54116587930155
// MI455X (gfx1250) — matrix operand order repaired and checked by reading, not compiled, not run on hardware
//
#include <hip/hip_runtime.h>
#include <cstdint>
#include <cstddef>

// ---------------- CDNA5 WMMA types ----------------
typedef _Float16 v16h __attribute__((ext_vector_type(16)));
typedef _Float16 v8h  __attribute__((ext_vector_type(8)));
typedef _Float16 v4h  __attribute__((ext_vector_type(4)));
typedef float    v8f  __attribute__((ext_vector_type(8)));

#define F_IN   75
#define KP1    96    // F_IN padded to 3 k-tiles of 32
#define HC1    512   // 8 heads x 64
#define NH1    8
#define C1     64
#define HC2    32    // 1 head x 32
#define C2     32
#define NEGSL  0.2f

__device__ __forceinline__ float leaky(float x) { return x < 0.0f ? NEGSL * x : x; }

// ================= f16 pre-packing =================
__global__ void cvt_x_f16(const float* __restrict__ x, _Float16* __restrict__ xh, int n) {
  long i = (long)blockIdx.x * blockDim.x + threadIdx.x;
  if (i >= (long)n * KP1) return;
  int row = (int)(i / KP1), k = (int)(i % KP1);
  xh[i] = (k < F_IN) ? (_Float16)x[(size_t)row * F_IN + k] : (_Float16)0.0f;
}
__global__ void cvt_w1t_f16(const float* __restrict__ W, _Float16* __restrict__ Wt) {
  int i = blockIdx.x * blockDim.x + threadIdx.x;          // 512*96
  if (i >= HC1 * KP1) return;
  int c = i / KP1, k = i % KP1;
  Wt[i] = (k < F_IN) ? (_Float16)W[(size_t)k * HC1 + c] : (_Float16)0.0f;
}
__global__ void cvt_w2t_f16(const float* __restrict__ W, _Float16* __restrict__ Wt) {
  int i = blockIdx.x * blockDim.x + threadIdx.x;          // 32*512
  if (i >= HC2 * HC1) return;
  int c = i / HC1, k = i % HC1;
  Wt[i] = (_Float16)W[(size_t)k * HC2 + c];
}

// pack two aligned 8-half loads into a v16h fragment
__device__ __forceinline__ v16h pack16(const _Float16* lo, const _Float16* hi8) {
  v8h a = *(const v8h*)lo;
  v8h b = *(const v8h*)hi8;
  return __builtin_shufflevector(a, b, 0,1,2,3,4,5,6,7,8,9,10,11,12,13,14,15);
}

// ---------------- GEMM1: h1[N,512] = xh[N,96] @ W1t'[96,512] ----------------
// grid = N/16 blocks, 256 threads (8 waves); wave w owns output cols [w*64, w*64+64)
__global__ __launch_bounds__(256) void gat_gemm1(const _Float16* __restrict__ xh,
                                                 const _Float16* __restrict__ W1t,
                                                 float* __restrict__ h1, int n) {
  const int lane    = threadIdx.x & 31;
  const int wave    = threadIdx.x >> 5;
  const int rowBase = blockIdx.x * 16;
  const int colBase = wave * 64;
  const int M       = lane & 15;
  const bool hi     = lane >= 16;
  int arow = rowBase + M; if (arow >= n) arow = n - 1;
  const _Float16* __restrict__ xrow = xh + (size_t)arow * KP1;

  v8f acc[4];
#pragma unroll
  for (int t = 0; t < 4; ++t) acc[t] = (v8f){0.f,0.f,0.f,0.f,0.f,0.f,0.f,0.f};

#pragma unroll
  for (int kt = 0; kt < 3; ++kt) {
    const int k0 = kt * 32;
    // A 16x32 f16 fragment: lanes0-15 halves = K{0..7,16..23}; lanes16-31 = K{8..15,24..31}
    v16h a = pack16(xrow + k0 + (hi ? 8 : 0), xrow + k0 + 16 + (hi ? 8 : 0));
#pragma unroll
    for (int t = 0; t < 4; ++t) {
      const _Float16* __restrict__ brow = W1t + (size_t)(colBase + t * 16 + M) * KP1;
      // B 32x16 fragment: lanes0-15 K=k0+{0..7,16..23}, lanes16-31 K=k0+{8..15,24..31} (W^T row)
      const _Float16* bp = brow + k0 + (hi ? 8 : 0);
      v16h b = pack16(bp, bp + 16);
      acc[t] = __builtin_amdgcn_wmma_f32_16x16x32_f16(false, a, false, b,
                                                      (short)0, acc[t], false, false);
    }
  }
#pragma unroll
  for (int t = 0; t < 4; ++t) {
    const int ncol = colBase + t * 16 + M;
#pragma unroll
    for (int r = 0; r < 8; ++r) {
      int orow = rowBase + r + (hi ? 8 : 0);
      if (orow < n) h1[(size_t)orow * HC1 + ncol] = acc[t][r];
    }
  }
}

// ---------------- GEMM2: h2[N,32] = h1act[N,512] @ W2t'[512,32] ----------------
// grid = N/16 blocks, 64 threads (2 waves); wave w owns cols [w*16, w*16+16)
__global__ __launch_bounds__(64) void gat_gemm2(const _Float16* __restrict__ act,
                                                const _Float16* __restrict__ W2t,
                                                float* __restrict__ h2, int n) {
  const int lane    = threadIdx.x & 31;
  const int wave    = threadIdx.x >> 5;
  const int rowBase = blockIdx.x * 16;
  const int M       = lane & 15;
  const bool hi     = lane >= 16;
  int arow = rowBase + M; if (arow >= n) arow = n - 1;
  const _Float16* __restrict__ hrow = act + (size_t)arow * HC1;
  const _Float16* __restrict__ brow = W2t + (size_t)(wave * 16 + M) * HC1;

  v8f acc = (v8f){0.f,0.f,0.f,0.f,0.f,0.f,0.f,0.f};
#pragma unroll 4
  for (int kt = 0; kt < 16; ++kt) {
    const int k0 = kt * 32;
    v16h a = pack16(hrow + k0 + (hi ? 8 : 0), hrow + k0 + 16 + (hi ? 8 : 0));
    const _Float16* bp = brow + k0 + (hi ? 8 : 0);
    v16h b = pack16(bp, bp + 16);
    acc = __builtin_amdgcn_wmma_f32_16x16x32_f16(false, a, false, b,
                                                 (short)0, acc, false, false);
  }
  const int ncol = wave * 16 + M;
#pragma unroll
  for (int r = 0; r < 8; ++r) {
    int orow = rowBase + r + (hi ? 8 : 0);
    if (orow < n) h2[(size_t)orow * HC2 + ncol] = acc[r];
  }
}

// ---------------- per-node attention scores ----------------
__global__ void gat_scores(const float* __restrict__ h, const float* __restrict__ a_src,
                           const float* __restrict__ a_dst, float* __restrict__ ssrc,
                           float* __restrict__ sdst, int n, int H, int C) {
  int idx = blockIdx.x * blockDim.x + threadIdx.x;
  if (idx >= n * H) return;
  int node = idx / H, head = idx % H;
  const float* __restrict__ hr = h + (size_t)node * H * C + head * C;
  const float* __restrict__ as = a_src + head * C;
  const float* __restrict__ ad = a_dst + head * C;
  float s0 = 0.f, s1 = 0.f;
  for (int c = 0; c < C; ++c) { float v = hr[c]; s0 += v * as[c]; s1 += v * ad[c]; }
  ssrc[idx] = s0; sdst[idx] = s1;
}

// ================= CSR-by-destination build (once, reused by both layers) =================
__device__ __forceinline__ void edge_sd(const int* __restrict__ ei, int E, int id,
                                        int& s, int& d) {
  if (id < E) { s = ei[id]; d = ei[E + id]; } else { s = d = id - E; }
}

__global__ void count_deg(const int* __restrict__ ei, int E, int n, int* __restrict__ deg) {
  int id = blockIdx.x * blockDim.x + threadIdx.x;
  if (id >= E + n) return;
  int s, d; edge_sd(ei, E, id, s, d);
  atomicAdd(&deg[d], 1);
}

// single-block exclusive scan (n=50000 here)
__global__ __launch_bounds__(1024) void scan_excl(const int* __restrict__ deg,
                                                  int* __restrict__ off,
                                                  int* __restrict__ cursor, int n) {
  __shared__ int buf[1024];
  __shared__ int carry_s;
  if (threadIdx.x == 0) carry_s = 0;
  __syncthreads();
  for (int base = 0; base < n; base += 1024) {
    int i = base + threadIdx.x;
    int v = (i < n) ? deg[i] : 0;
    buf[threadIdx.x] = v;
    __syncthreads();
    for (int o = 1; o < 1024; o <<= 1) {
      int t = (threadIdx.x >= o) ? buf[threadIdx.x - o] : 0;
      __syncthreads();
      buf[threadIdx.x] += t;
      __syncthreads();
    }
    int carry = carry_s;
    int excl = carry + buf[threadIdx.x] - v;
    if (i < n) { off[i] = excl; cursor[i] = excl; }
    __syncthreads();
    if (threadIdx.x == 0) carry_s = carry + buf[1023];
    __syncthreads();
  }
  if (threadIdx.x == 0) off[n] = carry_s;
}

__global__ void fill_csr(const int* __restrict__ ei, int E, int n,
                         int* __restrict__ cursor, int* __restrict__ csrc) {
  int id = blockIdx.x * blockDim.x + threadIdx.x;
  if (id >= E + n) return;
  int s, d; edge_sd(ei, E, id, s, d);
  int pos = atomicAdd(&cursor[d], 1);
  csrc[pos] = s;
}

// ================= layer-1 aggregation (gather, no float atomics) =================
// one wave per destination node: in-wave segment max / partition / weighted sum,
// float4 row gathers, fused bias + ELU + f16 vector store of the layer-2 GEMM input.
__global__ __launch_bounds__(256) void gat_agg1(const int* __restrict__ off,
    const int* __restrict__ csrc, const float* __restrict__ ssrc,
    const float* __restrict__ sdst, const float* __restrict__ h1,
    const float* __restrict__ bias, _Float16* __restrict__ h1act, int n) {
  const int lane = threadIdx.x & 31;
  const int d = (blockIdx.x * blockDim.x + threadIdx.x) >> 5;
  if (d >= n) return;
  const int beg = off[d], end = off[d + 1];

  float sd[8];
  { const float4* sp = (const float4*)(sdst + (size_t)d * NH1);
    float4 u0 = sp[0], u1 = sp[1];
    sd[0]=u0.x; sd[1]=u0.y; sd[2]=u0.z; sd[3]=u0.w;
    sd[4]=u1.x; sd[5]=u1.y; sd[6]=u1.z; sd[7]=u1.w; }

  // ---- per-head segment max ----
  float mx[8];
#pragma unroll
  for (int h = 0; h < 8; ++h) mx[h] = -3.0e38f;
  for (int p = beg + lane; p < end; p += 32) {
    int s = csrc[p];
    const float4* sp = (const float4*)(ssrc + (size_t)s * NH1);
    float4 u0 = sp[0], u1 = sp[1];
    float sv[8] = {u0.x,u0.y,u0.z,u0.w,u1.x,u1.y,u1.z,u1.w};
#pragma unroll
    for (int h = 0; h < 8; ++h) mx[h] = fmaxf(mx[h], leaky(sv[h] + sd[h]));
  }
#pragma unroll
  for (int h = 0; h < 8; ++h)
#pragma unroll
    for (int o = 16; o > 0; o >>= 1) mx[h] = fmaxf(mx[h], __shfl_xor(mx[h], o, 32));

  // ---- partition function ----
  float zz[8];
#pragma unroll
  for (int h = 0; h < 8; ++h) zz[h] = 0.f;
  for (int p = beg + lane; p < end; p += 32) {
    int s = csrc[p];
    const float4* sp = (const float4*)(ssrc + (size_t)s * NH1);
    float4 u0 = sp[0], u1 = sp[1];
    float sv[8] = {u0.x,u0.y,u0.z,u0.w,u1.x,u1.y,u1.z,u1.w};
#pragma unroll
    for (int h = 0; h < 8; ++h) zz[h] += __expf(leaky(sv[h] + sd[h]) - mx[h]);
  }
#pragma unroll
  for (int h = 0; h < 8; ++h)
#pragma unroll
    for (int o = 16; o > 0; o >>= 1) zz[h] += __shfl_xor(zz[h], o, 32);

  // ---- weighted gather of the full 512-float output row (float4 lanes) ----
  // float4 group q = it*32 + lane covers channels 4q..4q+3; head = q>>4 = it*2 + (lane>=16)
  float4 acc4[4];
#pragma unroll
  for (int it = 0; it < 4; ++it) acc4[it] = make_float4(0.f, 0.f, 0.f, 0.f);
  for (int p = beg; p < end; ++p) {
    int s = csrc[p];                                   // uniform -> broadcast load
    if (p + 1 < end) __builtin_prefetch(h1 + (size_t)csrc[p + 1] * HC1, 0, 0);
    float aval = 0.f;
    if (lane < 8)
      aval = __expf(leaky(ssrc[(size_t)s * NH1 + lane] + sd[lane]) - mx[lane])
             / (zz[lane] + 1e-16f);
    const float4* __restrict__ hrow = (const float4*)(h1 + (size_t)s * HC1);
#pragma unroll
    for (int it = 0; it < 4; ++it) {
      int q = it * 32 + lane;
      float al = __shfl(aval, q >> 4, 32);
      float4 hv = hrow[q];
      acc4[it].x += hv.x * al;
      acc4[it].y += hv.y * al;
      acc4[it].z += hv.z * al;
      acc4[it].w += hv.w * al;
    }
  }
  // fused bias + ELU + f16 convert (layer-2 GEMM input), 8B vector stores
#pragma unroll
  for (int it = 0; it < 4; ++it) {
    int q = it * 32 + lane;
    const float4* bp = (const float4*)(bias + 4 * q);
    float4 bv = *bp;
    float v0 = acc4[it].x + bv.x, v1 = acc4[it].y + bv.y;
    float v2 = acc4[it].z + bv.z, v3 = acc4[it].w + bv.w;
    v0 = v0 > 0.f ? v0 : __expf(v0) - 1.f;
    v1 = v1 > 0.f ? v1 : __expf(v1) - 1.f;
    v2 = v2 > 0.f ? v2 : __expf(v2) - 1.f;
    v3 = v3 > 0.f ? v3 : __expf(v3) - 1.f;
    v4h o; o[0] = (_Float16)v0; o[1] = (_Float16)v1; o[2] = (_Float16)v2; o[3] = (_Float16)v3;
    *(v4h*)(h1act + (size_t)d * HC1 + 4 * q) = o;
  }
}

// ================= layer-2 aggregation (H=1, C=32) =================
__global__ __launch_bounds__(256) void gat_agg2(const int* __restrict__ off,
    const int* __restrict__ csrc, const float* __restrict__ ssrc,
    const float* __restrict__ sdst, const float* __restrict__ h2,
    float* __restrict__ out, int n) {
  const int lane = threadIdx.x & 31;
  const int d = (blockIdx.x * blockDim.x + threadIdx.x) >> 5;
  if (d >= n) return;
  const int beg = off[d], end = off[d + 1];
  const float sdd = sdst[d];

  float mx = -3.0e38f;
  for (int p = beg + lane; p < end; p += 32) mx = fmaxf(mx, leaky(ssrc[csrc[p]] + sdd));
#pragma unroll
  for (int o = 16; o > 0; o >>= 1) mx = fmaxf(mx, __shfl_xor(mx, o, 32));

  float zz = 0.f;
  for (int p = beg + lane; p < end; p += 32) zz += __expf(leaky(ssrc[csrc[p]] + sdd) - mx);
#pragma unroll
  for (int o = 16; o > 0; o >>= 1) zz += __shfl_xor(zz, o, 32);

  float acc = 0.f;
  for (int p = beg; p < end; ++p) {
    int s = csrc[p];
    float alpha = __expf(leaky(ssrc[s] + sdd) - mx) / (zz + 1e-16f);
    acc += h2[(size_t)s * C2 + lane] * alpha;
  }
  out[(size_t)d * C2 + lane] = acc;                    // bias b2 folded into MLP
}

// ================= bias2 + MLP head =================
__global__ void gat_mlp(const float* __restrict__ h, const float* __restrict__ b2,
                        const float* __restrict__ lw1, const float* __restrict__ lb1,
                        const float* __restrict__ lw2, const float* __restrict__ lb2,
                        float* __restrict__ y, int n) {
  int i = blockIdx.x * blockDim.x + threadIdx.x;
  if (i >= n) return;
  float t[C2];
#pragma unroll
  for (int c = 0; c < C2; ++c) t[c] = h[(size_t)i * C2 + c] + b2[c];
  float acc = lb2[0];
#pragma unroll
  for (int k = 0; k < 16; ++k) {
    float u = lb1[k];
#pragma unroll
    for (int c = 0; c < C2; ++c) u += t[c] * lw1[c * 16 + k];
    acc += fmaxf(u, 0.f) * lw2[k];
  }
  y[i] = acc;
}

// ================= host orchestration =================
extern "C" void kernel_launch(void* const* d_in, const int* in_sizes, int n_in,
                              void* d_out, int out_size, void* d_ws, size_t ws_size,
                              hipStream_t stream) {
  const float* x     = (const float*)d_in[0];
  const float* W1    = (const float*)d_in[1];
  const float* asrc1 = (const float*)d_in[2];
  const float* adst1 = (const float*)d_in[3];
  const float* b1    = (const float*)d_in[4];
  const float* W2    = (const float*)d_in[5];
  const float* asrc2 = (const float*)d_in[6];
  const float* adst2 = (const float*)d_in[7];
  const float* b2    = (const float*)d_in[8];
  const float* lw1   = (const float*)d_in[9];
  const float* lb1   = (const float*)d_in[10];
  const float* lw2   = (const float*)d_in[11];
  const float* lb2   = (const float*)d_in[12];
  const int*   ei    = (const int*)d_in[13];

  const int n  = in_sizes[0] / F_IN;      // 50000
  const int E  = in_sizes[13] / 2;        // 800000
  const int ET = E + n;                   // with self-loops

  // workspace partition (256B aligned)
  char* ws = (char*)d_ws; size_t off_b = 0;
  auto take = [&](size_t bytes) -> void* {
    void* p = ws + off_b; off_b += (bytes + 255) & ~(size_t)255; return p;
  };
  _Float16* xh    = (_Float16*)take((size_t)n * KP1 * 2);
  _Float16* W1t   = (_Float16*)take((size_t)HC1 * KP1 * 2);
  _Float16* W2t   = (_Float16*)take((size_t)HC2 * HC1 * 2);
  float*    h1    = (float*)take((size_t)n * HC1 * 4);
  _Float16* h1act = (_Float16*)take((size_t)n * HC1 * 2);
  float*    ssrc1 = (float*)take((size_t)n * NH1 * 4);
  float*    sdst1 = (float*)take((size_t)n * NH1 * 4);
  float*    h2    = (float*)take((size_t)n * HC2 * 4);
  float*    out2  = (float*)take((size_t)n * HC2 * 4);
  float*    ssrc2 = (float*)take((size_t)n * 4);
  float*    sdst2 = (float*)take((size_t)n * 4);
  int*      deg   = (int*)take((size_t)n * 4);
  int*      offs  = (int*)take((size_t)(n + 1) * 4);
  int*      curs  = (int*)take((size_t)n * 4);
  int*      csrc  = (int*)take((size_t)ET * 4);
  (void)ws_size; (void)n_in; (void)out_size;

  const int rowTiles  = (n + 15) / 16;
  const int aggBlocks = (n + 7) / 8;                   // 8 waves / 256-thread block

  // ---- f16 pre-packing for WMMA feeds ----
  cvt_x_f16 <<<(unsigned)(((long)n * KP1 + 255) / 256), 256, 0, stream>>>(x, xh, n);
  cvt_w1t_f16<<<(HC1 * KP1 + 255) / 256, 256, 0, stream>>>(W1, W1t);
  cvt_w2t_f16<<<(HC2 * HC1 + 255) / 256, 256, 0, stream>>>(W2, W2t);

  // ---- CSR by destination (shared by both layers) ----
  hipMemsetAsync(deg, 0, (size_t)n * 4, stream);
  count_deg<<<(ET + 255) / 256, 256, 0, stream>>>(ei, E, n, deg);
  scan_excl<<<1, 1024, 0, stream>>>(deg, offs, curs, n);
  fill_csr <<<(ET + 255) / 256, 256, 0, stream>>>(ei, E, n, curs, csrc);

  // ---- layer 1 ----
  gat_gemm1<<<rowTiles, 256, 0, stream>>>(xh, W1t, h1, n);
  gat_scores<<<(n * NH1 + 255) / 256, 256, 0, stream>>>(h1, asrc1, adst1, ssrc1, sdst1, n, NH1, C1);
  gat_agg1<<<aggBlocks, 256, 0, stream>>>(offs, csrc, ssrc1, sdst1, h1, b1, h1act, n);

  // ---- layer 2 ----
  gat_gemm2<<<rowTiles, 64, 0, stream>>>(h1act, W2t, h2, n);
  gat_scores<<<(n + 255) / 256, 256, 0, stream>>>(h2, asrc2, adst2, ssrc2, sdst2, n, 1, C2);
  gat_agg2<<<aggBlocks, 256, 0, stream>>>(offs, csrc, ssrc2, sdst2, h2, out2, n);

  // ---- MLP head ----
  gat_mlp<<<(n + 255) / 256, 256, 0, stream>>>(out2, b2, lw1, lb1, lw2, lb2, (float*)d_out, n);
}
